// HashNetwork_62508954026128
// MI455X (gfx1250) — matrix operand order repaired and checked by reading, not compiled, not run on hardware
//
#include <hip/hip_runtime.h>

typedef _Float16 h16;
typedef __attribute__((ext_vector_type(16))) _Float16 v16h;
typedef __attribute__((ext_vector_type(8)))  _Float16 v8h;
typedef __attribute__((ext_vector_type(8)))  float    v8f;
typedef __attribute__((ext_vector_type(4)))  float    f32x4;

#define HDIM   1024
#define BDIM   8192
#define DIN    256
#define DOUT   256
#define NLAYER 64

#define BM 128
#define BN 256
#define BK 32
#define BKP 40          // padded LDS row stride (halves): 80B -> conflict-free b128
#define NTHREADS 256

// ---------------------------------------------------------------------------
// Transpose + affine: x [DIN][BDIM] f32  ->  XmT [BDIM][DIN] f16, val = 2x-1
// ---------------------------------------------------------------------------
__global__ __launch_bounds__(256) void xpose_kernel(const float* __restrict__ x,
                                                    h16* __restrict__ xmT) {
  __shared__ float tile[32][33];
  const int t  = threadIdx.x;
  const int k0 = blockIdx.y * 32;   // feature dim (DIN)
  const int n0 = blockIdx.x * 32;   // batch dim
  const int c  = t & 31;
  const int r  = t >> 5;            // 0..7
#pragma unroll
  for (int i = 0; i < 4; i++) {
    const int k = k0 + r + i * 8;
    tile[r + i * 8][c] = x[(size_t)k * BDIM + (n0 + c)];
  }
  __syncthreads();
#pragma unroll
  for (int i = 0; i < 4; i++) {
    const int n = n0 + r + i * 8;
    const float v = tile[c][r + i * 8];
    xmT[(size_t)n * DIN + (k0 + c)] = (h16)(2.0f * v - 1.0f);
  }
}

// ---------------------------------------------------------------------------
// Fused GEMM layer:  acc[m,n] = sum_k W[m,k] * BT[n,k]       (BT is K-major)
//   RES:    acc += BT[n,m]           (residual, requires M == KD)
//   clip:   r = clamp(acc, 0, 1)
//   !FINAL: OutT[n, m] = (h16)(2r-1)   (feeds next layer, K-major transposed)
//   FINAL:  OutF[m, n] = r             (fp32 reference output layout)
// Block tile 128x256, 8 waves (2M x 4N), wave tile 64x64 (4x4 WMMA), K-step 32.
// LDS rows padded to 40 halves (80B) -> all ds_*_b128 are bank-conflict-free.
// ---------------------------------------------------------------------------
template <int KD, bool RES, bool FINAL>
__global__ __launch_bounds__(NTHREADS)
void gemm_kernel(const float* __restrict__ W,    // [M][KD] f32
                 const h16* __restrict__ BT,     // [BDIM][KD] f16
                 h16* __restrict__ OutT,         // [BDIM][M] f16 (if !FINAL)
                 float* __restrict__ OutF) {     // [M][BDIM] f32 (if FINAL)
  __shared__ __align__(64) h16 As[2][BM][BKP];   // 20 KB
  __shared__ __align__(64) h16 Bs[2][BN][BKP];   // 40 KB

  const int t     = threadIdx.x;
  const int lane  = t & 31;
  const int wid   = t >> 5;          // 0..7
  const int waveM = wid >> 2;        // 0..1  (64 rows each)
  const int waveN = wid & 3;         // 0..3  (64 cols each)
  const int M0    = blockIdx.y * BM;
  const int N0    = blockIdx.x * BN;
  const int l16   = lane & 15;
  const int sel   = lane >> 4;       // 0/1 (which half of the wave)

  v8f c[4][4];
#pragma unroll
  for (int i = 0; i < 4; i++)
#pragma unroll
    for (int j = 0; j < 4; j++) c[i][j] = v8f{};

  // ---- staging: thread t loads A half-row (16 f32 -> f16) and one full B row
  const int ra = t >> 1;             // 0..127 : A row within block tile
  const int kh = (t & 1) * 16;       // 0 / 16 : which 16-wide K chunk
  const int rb = t;                  // 0..255 : B row (n) within block tile

  const float* aptr = W  + (size_t)(M0 + ra) * KD + kh;
  const h16*   bptr = BT + (size_t)(N0 + rb) * KD;

  f32x4 aS[4];
  v8h   bS[4];

#pragma unroll
  for (int q = 0; q < 4; q++) aS[q] = *(const f32x4*)(aptr + q * 4);
#pragma unroll
  for (int q = 0; q < 4; q++) bS[q] = *(const v8h*)(bptr + q * 8);

  auto stage = [&](int s) {
    v8h p0, p1;
#pragma unroll
    for (int e = 0; e < 4; e++) {
      p0[e]     = (h16)aS[0][e];
      p0[4 + e] = (h16)aS[1][e];
      p1[e]     = (h16)aS[2][e];
      p1[4 + e] = (h16)aS[3][e];
    }
    *(v8h*)&As[s][ra][kh]     = p0;
    *(v8h*)&As[s][ra][kh + 8] = p1;
#pragma unroll
    for (int q = 0; q < 4; q++) *(v8h*)&Bs[s][rb][q * 8] = bS[q];
  };

  stage(0);
  __syncthreads();

  const int nk = KD / BK;
  for (int ks = 0; ks < nk; ks++) {
    const int s = ks & 1;
    if (ks + 1 < nk) {               // prefetch next tile into registers
      const int koff = (ks + 1) * BK;
#pragma unroll
      for (int q = 0; q < 4; q++) aS[q] = *(const f32x4*)(aptr + koff + q * 4);
#pragma unroll
      for (int q = 0; q < 4; q++) bS[q] = *(const v8h*)(bptr + koff + q * 8);
    }

    // ---- A fragments: lane l16 = row, K chunks {sel*8..+7} and {sel*8+16..+23}
    v16h a[4], b[4];
#pragma unroll
    for (int i = 0; i < 4; i++) {
      const h16* p = &As[s][waveM * 64 + i * 16 + l16][sel * 8];
      v8h lo = *(const v8h*)p;
      v8h hi = *(const v8h*)(p + 16);
      a[i] = __builtin_shufflevector(lo, hi, 0, 1, 2, 3, 4, 5, 6, 7,
                                             8, 9, 10, 11, 12, 13, 14, 15);
    }
    // ---- B fragments: lane l16 = column n, K chunks {sel*8..+7} and {sel*8+16..+23}
#pragma unroll
    for (int j = 0; j < 4; j++) {
      const h16* p = &Bs[s][waveN * 64 + j * 16 + l16][sel * 8];
      v8h lo = *(const v8h*)p;
      v8h hi = *(const v8h*)(p + 16);
      b[j] = __builtin_shufflevector(lo, hi, 0, 1, 2, 3, 4, 5, 6, 7,
                                             8, 9, 10, 11, 12, 13, 14, 15);
    }

#pragma unroll
    for (int i = 0; i < 4; i++)
#pragma unroll
      for (int j = 0; j < 4; j++)
        c[i][j] = __builtin_amdgcn_wmma_f32_16x16x32_f16(
            false, a[i], false, b[j], (short)0, c[i][j], false, false);

    if (ks + 1 < nk) {
      __syncthreads();               // everyone finished reading buf s^1
      stage(s ^ 1);
      __syncthreads();               // staged data visible before next compute
    }
  }

  // ---- epilogue: D layout = lane l16 -> column n, VGPR v -> row m = sel*8+v
#pragma unroll
  for (int i = 0; i < 4; i++) {
#pragma unroll
    for (int j = 0; j < 4; j++) {
      const int n = N0 + waveN * 64 + j * 16 + l16;
      const int m = M0 + waveM * 64 + i * 16 + sel * 8;
      v8f acc = c[i][j];
      if (RES) {                     // residual: + hm[m,n] == BT[n][m] (M==KD)
        v8h r = *(const v8h*)(BT + (size_t)n * KD + m);
#pragma unroll
        for (int v = 0; v < 8; v++) acc[v] += (float)r[v];
      }
#pragma unroll
      for (int v = 0; v < 8; v++) acc[v] = fminf(fmaxf(acc[v], 0.0f), 1.0f);
      if (FINAL) {
#pragma unroll
        for (int v = 0; v < 8; v++)
          OutF[(size_t)(m + v) * BDIM + n] = acc[v];
      } else {
        v8h o;
#pragma unroll
        for (int v = 0; v < 8; v++) o[v] = (h16)(2.0f * acc[v] - 1.0f);
        *(v8h*)(OutT + (size_t)n * HDIM + m) = o;   // one 16B store
      }
    }
  }
}

// ---------------------------------------------------------------------------
extern "C" void kernel_launch(void* const* d_in, const int* in_sizes, int n_in,
                              void* d_out, int out_size, void* d_ws,
                              size_t ws_size, hipStream_t stream) {
  (void)in_sizes; (void)n_in; (void)out_size; (void)ws_size;

  const float* x           = (const float*)d_in[0];  // [256][8192]
  const float* expansion   = (const float*)d_in[1];  // [1024][256]
  const float* hidden      = (const float*)d_in[2];  // [64][1024][1024]
  const float* compression = (const float*)d_in[3];  // [256][1024]
  float* out = (float*)d_out;                        // [256][8192]

  char* ws = (char*)d_ws;
  h16* xmT  = (h16*)ws;                                // 8192*256*2  =  4 MB
  h16* buf0 = (h16*)(ws + (size_t)4  * 1024 * 1024);   // 8192*1024*2 = 16 MB
  h16* buf1 = (h16*)(ws + (size_t)20 * 1024 * 1024);   // 16 MB

  // 1) transpose + (2x-1) convert of input
  dim3 tb(256), tg(BDIM / 32, DIN / 32);
  xpose_kernel<<<tg, tb, 0, stream>>>(x, xmT);

  // 2) expansion:  hm1 = 2*clip(E @ xm) - 1
  dim3 gb(NTHREADS);
  dim3 g1(BDIM / BN, HDIM / BM);
  gemm_kernel<DIN, false, false><<<g1, gb, 0, stream>>>(expansion, xmT, buf0, nullptr);

  // 3) 64 hidden layers: hm' = 2*clip(W @ hm + hm) - 1   (ping-pong buffers)
  h16* pin = buf0; h16* pout = buf1;
  for (int l = 0; l < NLAYER; l++) {
    gemm_kernel<HDIM, true, false><<<g1, gb, 0, stream>>>(
        hidden + (size_t)l * HDIM * HDIM, pin, pout, nullptr);
    h16* tmp = pin; pin = pout; pout = tmp;
  }

  // 4) compression: out = clip(C @ hm), fp32 [256][8192]
  dim3 g2(BDIM / BN, DOUT / BM);
  gemm_kernel<HDIM, false, true><<<g2, gb, 0, stream>>>(compression, pin, nullptr, out);
}
